// AttnGate_5712306504201
// MI455X (gfx1250) — matrix operand order repaired and checked by reading, not compiled, not run on hardware
//
#include <hip/hip_runtime.h>
#include <hip/hip_bf16.h>

// ---------------- types for WMMA bf16 ----------------
typedef __attribute__((ext_vector_type(16))) __bf16 v16bf;
typedef __attribute__((ext_vector_type(8)))  float  v8f;

union Frag {
    v16bf v;
    uint4 q[2];
};

__device__ __forceinline__ unsigned short f2bf(float f) {
    union { float f; unsigned u; } x; x.f = f;
    unsigned r = x.u + 0x7FFFu + ((x.u >> 16) & 1u);   // round-to-nearest-even
    return (unsigned short)(r >> 16);
}

// ---------------- constants ----------------
#define BB   2
#define HQ   32
#define HK   8
#define SS   8192
#define DD   128
#define HID  128
#define BSZ  16
#define NBL  512            // S / BS
#define C2D  256            // 2*D
#define LSTR 516            // LDS row stride (floats): 16B-aligned, breaks 2-way conflict

// =====================================================================
// Kernel 1: block pooling  (B,H,S,D) f32 -> (B,H,NB,2D) bf16 [mean | max]
// grid.x = B*H*NB/4, block = 128 = 4 waves; wave `sub` handles one nb block,
// lane covers a float4 of D. Each q/k element read exactly once via b128.
// =====================================================================
__global__ void __launch_bounds__(128) pool_kernel(const float* __restrict__ x,
                                                   unsigned short* __restrict__ pooled) {
    int lane = threadIdx.x & 31;
    int sub  = threadIdx.x >> 5;                     // 0..3
    size_t idx = (size_t)blockIdx.x * 4 + sub;       // (b*H + h)*NB + nb
    int d4 = lane * 4;
    const float* src = x + idx * (size_t)(BSZ * DD) + d4;

    float4 s = {0.f, 0.f, 0.f, 0.f};
    float4 m = {-3.0e38f, -3.0e38f, -3.0e38f, -3.0e38f};
#pragma unroll
    for (int i = 0; i < BSZ; ++i) {
        float4 v = *(const float4*)(src + i * DD);   // global_load_b128
        s.x += v.x; s.y += v.y; s.z += v.z; s.w += v.w;
        m.x = fmaxf(m.x, v.x); m.y = fmaxf(m.y, v.y);
        m.z = fmaxf(m.z, v.z); m.w = fmaxf(m.w, v.w);
    }
    ushort4 mean_o, max_o;
    mean_o.x = f2bf(s.x * 0.0625f); mean_o.y = f2bf(s.y * 0.0625f);
    mean_o.z = f2bf(s.z * 0.0625f); mean_o.w = f2bf(s.w * 0.0625f);
    max_o.x  = f2bf(m.x); max_o.y = f2bf(m.y);
    max_o.z  = f2bf(m.z); max_o.w = f2bf(m.w);
    *(ushort4*)(pooled + idx * C2D + d4)      = mean_o;   // b64 store
    *(ushort4*)(pooled + idx * C2D + DD + d4) = max_o;    // b64 store
}

// =====================================================================
// Kernel 2: weight transpose+convert: (H, 2D, HID) f32 -> (H, HID, 2D) bf16
// =====================================================================
__global__ void __launch_bounds__(256) wt_kernel(const float* __restrict__ w,
                                                 unsigned short* __restrict__ wT,
                                                 int total) {
    int o = blockIdx.x * 256 + threadIdx.x;
    if (o >= total) return;
    int h   = o >> 15;            // / (HID*C2D) = /32768
    int rem = o & 32767;
    int n   = rem >> 8;           // / C2D
    int c   = rem & 255;
    wT[o] = f2bf(w[((size_t)h * C2D + c) * HID + n]);
}

// =====================================================================
// Kernel 3: projection GEMM (bf16 WMMA) + RoPE.
//   pooled (bh, NB, 2D) bf16  @  wT (h, HID, 2D) bf16  -> out (bh, NB, HID) bf16
// grid.x = B*H*32 (mtile), block = 128 (4 waves). Wave j computes N-tiles
// j and j+4 (RoPE pairs column n with n+64).
// =====================================================================
__global__ void __launch_bounds__(128) proj_rope_kernel(const unsigned short* __restrict__ pooled,
                                                        const unsigned short* __restrict__ wT,
                                                        const float* __restrict__ cosp,
                                                        const float* __restrict__ sinp,
                                                        unsigned short* __restrict__ outp,
                                                        int H) {
    int lane  = threadIdx.x & 31;
    int wave  = threadIdx.x >> 5;           // j = 0..3
    int mtile = blockIdx.x & 31;
    int bh    = blockIdx.x >> 5;            // b*H + h
    int b     = bh / H;
    int h     = bh - b * H;

    int l15   = lane & 15;
    int lhalf = lane >> 4;

    // A fragment base: 16x32 bf16 tile of pooled, row-major, K-stride 256
    const unsigned short* A = pooled + ((size_t)bh * NBL + mtile * 16 + l15) * C2D + lhalf * 16;
    // B fragments from transposed weights (HID x 2D row-major): per-lane rows
    const unsigned short* B0 = wT + ((size_t)h * HID + wave * 16 + l15) * C2D + lhalf * 16;
    const unsigned short* B1 = B0 + 64 * C2D;      // n-tile j+4 (= +64 columns)

    v8f acc0 = {};
    v8f acc1 = {};
#pragma unroll
    for (int kb = 0; kb < C2D; kb += 32) {
        Frag a, p0, p1;
        a.q[0]  = *(const uint4*)(A + kb);
        a.q[1]  = *(const uint4*)(A + kb + 8);
        p0.q[0] = *(const uint4*)(B0 + kb);
        p0.q[1] = *(const uint4*)(B0 + kb + 8);
        p1.q[0] = *(const uint4*)(B1 + kb);
        p1.q[1] = *(const uint4*)(B1 + kb + 8);
        acc0 = __builtin_amdgcn_wmma_f32_16x16x32_bf16(false, a.v, false, p0.v,
                                                       (short)0, acc0, false, false);
        acc1 = __builtin_amdgcn_wmma_f32_16x16x32_bf16(false, a.v, false, p1.v,
                                                       (short)0, acc1, false, false);
    }

    // RoPE epilogue on f32 accumulators. C/D layout: lane0-15 -> M=v, lane16-31 -> M=v+8.
    int col0 = wave * 16 + l15;       // < 64
    int col1 = col0 + 64;
    size_t cbase = (size_t)b * NBL * HID;
    size_t obase = (size_t)bh * NBL * HID;
#pragma unroll
    for (int v = 0; v < 8; ++v) {
        int ml  = v + (lhalf << 3);
        int nbi = mtile * 16 + ml;           // block position (RoPE "pos")
        size_t rb = (size_t)nbi * HID;
        float c0 = cosp[cbase + rb + col0];
        float s0 = sinp[cbase + rb + col0];
        float c1 = cosp[cbase + rb + col1];
        float s1 = sinp[cbase + rb + col1];
        float x0 = acc0[v], x1 = acc1[v];
        // out[n<64]  = x*cos - x[n+64]*sin ; out[n>=64] = x*cos + x[n-64]*sin
        outp[obase + rb + col0] = f2bf(x0 * c0 - x1 * s0);
        outp[obase + rb + col1] = f2bf(x1 * c1 + x0 * s1);
    }
}

// =====================================================================
// Kernel 4: scores (bf16 WMMA) + block-causal mask + fused softmax.
// grid.x = B*HQ*32 (mtile strip of 16 rows), block = 256 (8 waves).
// Wave w computes column tiles ct = w, w+8, ... with ct <= mtile (uniform
// per wave -> EXEC stays all-ones for WMMA). Scores live in an LDS strip;
// softmax reductions use float4 LDS reads + shfl_xor; output is b128 stores.
// =====================================================================
__global__ void __launch_bounds__(256) attn_kernel(const unsigned short* __restrict__ qh,
                                                   const unsigned short* __restrict__ kh,
                                                   float* __restrict__ out) {
    __shared__ float strip[16 * LSTR];

    int lane  = threadIdx.x & 31;
    int wave  = threadIdx.x >> 5;
    int mtile = blockIdx.x & 31;
    int bh    = blockIdx.x >> 5;         // b*HQ + hq
    int b     = bh >> 5;
    int hq    = bh & 31;
    int hk    = hq >> 2;                 // GQA: repeat factor HQ/HK = 4

    int l15   = lane & 15;
    int lhalf = lane >> 4;

    const unsigned short* A  = qh + ((size_t)bh * NBL + mtile * 16 + l15) * HID + lhalf * 16;
    const unsigned short* Kb = kh + ((size_t)(b * HK + hk) * NBL) * HID;

    // hoist the 4 A k-fragments (only 128-deep K)
    Frag afr[4];
#pragma unroll
    for (int kk = 0; kk < 4; ++kk) {
        afr[kk].q[0] = *(const uint4*)(A + kk * 32);
        afr[kk].q[1] = *(const uint4*)(A + kk * 32 + 8);
    }

    const float scale = 0.08838834764831845f;   // 1/sqrt(128)
    for (int ct = wave; ct <= mtile; ct += 8) {
        const unsigned short* Bp = Kb + ((size_t)(ct * 16 + l15)) * HID + lhalf * 16;
        v8f acc = {};
#pragma unroll
        for (int kk = 0; kk < 4; ++kk) {
            Frag bb;
            bb.q[0] = *(const uint4*)(Bp + kk * 32);
            bb.q[1] = *(const uint4*)(Bp + kk * 32 + 8);
            acc = __builtin_amdgcn_wmma_f32_16x16x32_bf16(false, afr[kk].v, false, bb.v,
                                                          (short)0, acc, false, false);
        }
#pragma unroll
        for (int v = 0; v < 8; ++v) {
            int ml = v + (lhalf << 3);
            float val = acc[v] * scale;
            if (ct == mtile && l15 > ml) val = -1.0e9f;   // triangular mask inside diag tile
            strip[ml * LSTR + ct * 16 + l15] = val;
        }
    }
    __syncthreads();

    // softmax: 2 rows per wave; float4 LDS reads, shfl_xor wave reductions
    int ncols = (mtile + 1) * 16;        // multiple of 16 -> quads never straddle
    for (int r = wave * 2; r < wave * 2 + 2; ++r) {
        const float* row = strip + r * LSTR;
        float mx = -3.0e38f;
        for (int c = lane * 4; c < ncols; c += 128) {
            float4 v = *(const float4*)(row + c);        // ds_load_b128
            mx = fmaxf(mx, fmaxf(fmaxf(v.x, v.y), fmaxf(v.z, v.w)));
        }
#pragma unroll
        for (int o = 16; o; o >>= 1) mx = fmaxf(mx, __shfl_xor(mx, o, 32));
        float sum = 0.f;
        for (int c = lane * 4; c < ncols; c += 128) {
            float4 v = *(const float4*)(row + c);
            sum += __expf(v.x - mx) + __expf(v.y - mx) +
                   __expf(v.z - mx) + __expf(v.w - mx);
        }
#pragma unroll
        for (int o = 16; o; o >>= 1) sum += __shfl_xor(sum, o, 32);
        float inv = 1.0f / sum;
        size_t ob = ((size_t)bh * NBL + mtile * 16 + r) * NBL;
        for (int c = lane * 4; c < NBL; c += 128) {
            float4 o4;
            if (c < ncols) {                              // whole quad in-bounds
                float4 v = *(const float4*)(row + c);
                o4.x = __expf(v.x - mx) * inv;
                o4.y = __expf(v.y - mx) * inv;
                o4.z = __expf(v.z - mx) * inv;
                o4.w = __expf(v.w - mx) * inv;
            } else {
                o4.x = o4.y = o4.z = o4.w = 0.0f;         // masked region
            }
            *(float4*)(out + ob + c) = o4;                // global_store_b128
        }
    }
}

// =====================================================================
// launch
// =====================================================================
extern "C" void kernel_launch(void* const* d_in, const int* in_sizes, int n_in,
                              void* d_out, int out_size, void* d_ws, size_t ws_size,
                              hipStream_t stream) {
    const float* q    = (const float*)d_in[0];   // (2,32,8192,128)
    const float* k    = (const float*)d_in[1];   // (2,8,8192,128)
    /* d_in[2]: attention_mask — block-causal tril, handled analytically */
    const float* cosp = (const float*)d_in[3];   // (2,512,128)
    const float* sinp = (const float*)d_in[4];   // (2,512,128)
    const float* wq   = (const float*)d_in[5];   // (32,256,128)
    const float* wk   = (const float*)d_in[6];   // (8,256,128)

    char* ws = (char*)d_ws;
    unsigned short* pq  = (unsigned short*)(ws + 0);          // 16 MB
    unsigned short* pk  = (unsigned short*)(ws + 16777216);   //  4 MB
    unsigned short* wqT = (unsigned short*)(ws + 20971520);   //  2 MB
    unsigned short* wkT = (unsigned short*)(ws + 23068672);   // .5 MB
    unsigned short* qh  = (unsigned short*)(ws + 23592960);   //  8 MB
    unsigned short* kh  = (unsigned short*)(ws + 31981568);   //  2 MB

    // 1) streaming pools (bandwidth-dominant stage, b128 loads)
    pool_kernel<<<BB * HQ * NBL / 4, 128, 0, stream>>>(q, pq);
    pool_kernel<<<BB * HK * NBL / 4, 128, 0, stream>>>(k, pk);

    // 2) weight transpose + bf16 convert (tiny)
    wt_kernel<<<(HQ * C2D * HID + 255) / 256, 256, 0, stream>>>(wq, wqT, HQ * C2D * HID);
    wt_kernel<<<(HK * C2D * HID + 255) / 256, 256, 0, stream>>>(wk, wkT, HK * C2D * HID);

    // 3) projection GEMMs + RoPE (WMMA bf16)
    proj_rope_kernel<<<BB * HQ * 32, 128, 0, stream>>>(pq, wqT, cosp, sinp, qh, HQ);
    proj_rope_kernel<<<BB * HK * 32, 128, 0, stream>>>(pk, wkT, cosp, sinp, kh, HK);

    // 4) QK^T (WMMA bf16, causal tile skipping) + fused softmax
    attn_kernel<<<BB * HQ * 32, 256, 0, stream>>>(qh, kh, (float*)d_out);
}
